// DeformableConv2dLayer_20349555048916
// MI455X (gfx1250) — matrix operand order repaired and checked by reading, not compiled, not run on hardware
//
#include <hip/hip_runtime.h>
#include <hip/hip_bf16.h>

typedef _Float16 v16h __attribute__((ext_vector_type(16)));
typedef _Float16 v8h  __attribute__((ext_vector_type(8)));
typedef float    v8f  __attribute__((ext_vector_type(8)));

#define HDIM 128
#define WDIM 128
#define CDIM 32
#define CODIM 32
#define NPTS 9
#define KDIM (NPTS * CDIM)   // 288
#define KP   296             // padded K stride (halves); 296*2=592 B = 37*16 -> keeps v8h 16B-aligned
#define WAVES 4

__global__ __launch_bounds__(32 * WAVES)
void deform_conv_wmma_kernel(const float* __restrict__ x,
                             const float* __restrict__ offs,
                             const float* __restrict__ Wg,
                             const float* __restrict__ bias,
                             float* __restrict__ out)
{
    __shared__ __align__(16) _Float16 sWt[CODIM * KP];          // Wt[cout][k]  (k = n*32+c)
    __shared__ __align__(16) _Float16 sA[WAVES][16 * KP];       // A[p][k] per wave

    const int tid  = threadIdx.x;
    const int lane = tid & 31;
    const int wave = tid >> 5;

    // block -> (b, h, w-segment of 64)
    const int blk  = blockIdx.x;
    const int wseg = blk & 1;             // WDIM/64 == 2 segments
    const int bh   = blk >> 1;
    const int h    = bh & (HDIM - 1);
    const int b    = bh >> 7;             // / HDIM
    const int w0   = wseg * 64 + wave * 16;

    // ---- Phase 0: weights (N,C,Cout) f32 -> LDS transposed f16: Wt[o][k] ----
    for (int idx = tid; idx < KDIM * CODIM; idx += blockDim.x) {
        const int k = idx >> 5;           // n*32+c
        const int o = idx & 31;
        sWt[o * KP + k] = (_Float16)Wg[idx];   // Wg flat index = k*32 + o = idx
    }

    // ---- Phase 1: bilinear sampling -> A[p][n*32+c] (lane == channel c) ----
    // Faithful TF-quirk init_off from stack(meshgrid).reshape(-1,2):
    const int di[9] = {0, 0, 1, 2, 2, 1, 0, 2, 1};
    const int dj[9] = {0, 1, 1, 2, 0, 2, 1, 0, 2};

    _Float16* Aw = sA[wave];
    const float* xb = x + (size_t)b * HDIM * WDIM * CDIM;

    for (int p = 0; p < 16; ++p) {
        const int wp = w0 + p;
        const float* opix = offs + ((size_t)((b * HDIM + h) * WDIM + wp)) * (2 * NPTS);
        const float offv = (lane < 2 * NPTS) ? opix[lane] : 0.0f;
#pragma unroll
        for (int n = 0; n < NPTS; ++n) {
            const float o0 = __shfl(offv, 2 * n, 32);
            const float o1 = __shfl(offv, 2 * n + 1, 32);
            float c0 = (float)(h  - 1 + di[n]) + o0;
            float c1 = (float)(wp - 1 + dj[n]) + o1;
            c0 = fminf(fmaxf(c0, 0.0f), (float)(HDIM - 1));
            c1 = fminf(fmaxf(c1, 0.0f), (float)(WDIM - 1));
            const float flt0 = floorf(c0), flt1 = floorf(c1);
            const int lt0 = (int)flt0,      lt1 = (int)flt1;
            const int rb0 = (int)ceilf(c0), rb1 = (int)ceilf(c1);
            const float f0 = c0 - flt0, f1 = c1 - flt1;

            const float v_lt = xb[((lt0 * WDIM) + lt1) * CDIM + lane];
            const float v_rb = xb[((rb0 * WDIM) + rb1) * CDIM + lane];
            const float v_lb = xb[((lt0 * WDIM) + rb1) * CDIM + lane];
            const float v_rt = xb[((rb0 * WDIM) + lt1) * CDIM + lane];

            const float v_t = v_lt + (v_rt - v_lt) * f0;
            const float v_b = v_lb + (v_rb - v_lb) * f0;
            const float m   = v_t + (v_b - v_t) * f1;

            Aw[p * KP + n * 32 + lane] = (_Float16)m;
        }
    }
    __syncthreads();

    // ---- Phase 2: D = A(16x288) x Wt^T(288x32) via v_wmma_f32_16x16x32_f16 ----
    v8f acc0 = {};
    v8f acc1 = {};
    const int mrow = lane & 15;
    const int aoff = (lane < 16) ? 0 : 16;   // A: lanes16-31 hold K=16..31
    const int boff = (lane < 16) ? 0 : 16;   // B: lanes16-31 hold K=16..31

#pragma unroll
    for (int kb = 0; kb < NPTS; ++kb) {      // K blocks of 32 == kernel point n
        // A fragment: 16 contiguous K values starting at aoff (matches B)
        const _Float16* ap = Aw + mrow * KP + kb * 32 + aoff;
        const v8h alo = *(const v8h*)ap;
        const v8h ahi = *(const v8h*)(ap + 8);
        v16h a;
#pragma unroll
        for (int i = 0; i < 8; ++i) { a[i] = alo[i]; a[i + 8] = ahi[i]; }

        // B fragments: lane holds column N=mrow, 16 contiguous K values starting at boff
        const _Float16* bp0 = sWt + mrow * KP        + kb * 32 + boff;  // cout 0..15
        const _Float16* bp1 = sWt + (mrow + 16) * KP + kb * 32 + boff;  // cout 16..31
        const v8h t0 = *(const v8h*)bp0, t1 = *(const v8h*)(bp0 + 8);
        const v8h u0 = *(const v8h*)bp1, u1 = *(const v8h*)(bp1 + 8);
        v16h b0, b1;
#pragma unroll
        for (int i = 0; i < 8; ++i) { b0[i] = t0[i]; b0[i + 8] = t1[i]; }
#pragma unroll
        for (int i = 0; i < 8; ++i) { b1[i] = u0[i]; b1[i + 8] = u1[i]; }

        acc0 = __builtin_amdgcn_wmma_f32_16x16x32_f16(false, a, false, b0,
                                                      (short)0, acc0, false, false);
        acc1 = __builtin_amdgcn_wmma_f32_16x16x32_f16(false, a, false, b1,
                                                      (short)0, acc1, false, false);
    }

    // ---- Phase 3: store C/D (lanes 0-15: M=r, N=lane; lanes 16-31: M=r+8, N=lane-16) ----
    const int co   = lane & 15;
    const int moff = (lane < 16) ? 0 : 8;
    const float bias0 = bias[co];
    const float bias1 = bias[co + 16];
#pragma unroll
    for (int r = 0; r < 8; ++r) {
        const int p = r + moff;
        float* orow = out + ((size_t)((b * HDIM + h) * WDIM + (w0 + p))) * CODIM;
        orow[co]      = acc0[r] + bias0;
        orow[co + 16] = acc1[r] + bias1;
    }
}

extern "C" void kernel_launch(void* const* d_in, const int* in_sizes, int n_in,
                              void* d_out, int out_size, void* d_ws, size_t ws_size,
                              hipStream_t stream) {
    const float* x    = (const float*)d_in[0];
    const float* offs = (const float*)d_in[1];
    const float* Wg   = (const float*)d_in[2];
    const float* bias = (const float*)d_in[3];
    float* out = (float*)d_out;

    const int B = in_sizes[0] / (HDIM * WDIM * CDIM);   // 8 for the reference setup
    const int blocks = B * HDIM * (WDIM / (16 * WAVES)); // each block: 4 waves x 16 pixels
    deform_conv_wmma_kernel<<<blocks, 32 * WAVES, 0, stream>>>(x, offs, Wg, bias, out);
}
